// Attention_9861244912350
// MI455X (gfx1250) — matrix operand order repaired and checked by reading, not compiled, not run on hardware
//
#include <hip/hip_runtime.h>
#include <hip/hip_bf16.h>

// ---------------------------------------------------------------------------
// MI455X (gfx1250) fused block-causal attention:
//   LN -> QKV GEMM (WMMA f16) -> flash attention (WMMA f16) -> out GEMM (WMMA)
// Matrix math: v_wmma_f32_16x16x32_f16 (wave32).
// Data movement: global_load_async_to_lds_b128 (ASYNCcnt) for GEMM staging,
// Tensor Data Mover (tensor_load_to_lds, TENSORcnt) for attention V tiles.
// ---------------------------------------------------------------------------

typedef __attribute__((ext_vector_type(16))) _Float16 v16h;
typedef __attribute__((ext_vector_type(8)))  _Float16 h8;
typedef __attribute__((ext_vector_type(8)))  float    v8f;
typedef __attribute__((ext_vector_type(4)))  unsigned u32x4;
typedef __attribute__((ext_vector_type(8)))  unsigned u32x8;

union V16 { v16h v; h8 h2[2]; };
union G0u { u32x4 v; unsigned u[4]; };
union G1u { u32x8 v; unsigned u[8]; };

static constexpr int Bc = 4;
static constexpr int Sc = 2048;
static constexpr int Dc = 1024;
static constexpr int Hc = 16;
static constexpr int Dh = 64;
static constexpr int INNER = Hc * Dh;   // 1024
static constexpr int Mrows = Bc * Sc;   // 8192

// ---- WMMA helper -----------------------------------------------------------
__device__ __forceinline__ v8f wmma_f16(v16h a, v16h b, v8f c) {
    return __builtin_amdgcn_wmma_f32_16x16x32_f16(
        /*neg_a=*/false, a, /*neg_b=*/false, b,
        /*c_mod=*/(short)0, c, /*reuse_a=*/false, /*reuse_b=*/false);
}

// Async global->LDS 16B copy (ASYNCcnt tracked).  VDST = LDS byte address,
// VADDR = 64-bit global address (generic shared ptr low 32 bits == LDS addr).
__device__ __forceinline__ void async_cp16(void* lds, const void* g) {
    asm volatile("global_load_async_to_lds_b128 %0, %1, off"
                 :: "v"((unsigned)(uintptr_t)lds),
                    "v"((unsigned long long)(uintptr_t)g)
                 : "memory");
}
__device__ __forceinline__ void wait_async0() {
    asm volatile("s_wait_asynccnt 0x0" ::: "memory");
}

// A fragment (16x32 f16): lane holds row m=lane&15; halves 0..7 -> k=khi..khi+7,
// halves 8..15 -> k=khi+16..khi+23, khi = 8*(lane>=16).  (ISA 7.12.2)
__device__ __forceinline__ v16h load_a(const _Float16* base, int stride) {
    int lane = threadIdx.x & 31;
    int m    = lane & 15;
    int khi  = (lane & 16) ? 8 : 0;
    const _Float16* p = base + m * stride + khi;
    V16 r;
    r.h2[0] = *(const h8*)(p);
    r.h2[1] = *(const h8*)(p + 16);
    return r.v;
}

// B fragment (32x16 f16): lane holds col n=lane&15; halves 0..7 -> k=khi..khi+7,
// halves 8..15 -> k=khi+16..khi+23, khi = 8*(lane>=16).
__device__ __forceinline__ v16h load_b(const _Float16* base, int stride) {
    int lane = threadIdx.x & 31;
    int khi  = (lane & 16) ? 8 : 0;
    const _Float16* p = base + khi * stride + (lane & 15);
    V16 r;
    #pragma unroll
    for (int s = 0; s < 8; ++s) {
        r.v[s]     = p[s * stride];
        r.v[8 + s] = p[(16 + s) * stride];
    }
    return r.v;
}

// ---- LayerNorm + f32->f16 --------------------------------------------------
__global__ void ln_f16_kernel(const float* __restrict__ x,
                              const float* __restrict__ gamma,
                              const float* __restrict__ beta,
                              _Float16* __restrict__ xn) {
    int row = blockIdx.x;
    int t   = threadIdx.x;
    const float* xr = x + (size_t)row * Dc;
    float s = 0.f, s2 = 0.f;
    for (int i = t; i < Dc; i += 256) { float v = xr[i]; s += v; s2 += v * v; }
    __shared__ float rs[256], rq[256];
    rs[t] = s; rq[t] = s2;
    __syncthreads();
    for (int st = 128; st > 0; st >>= 1) {
        if (t < st) { rs[t] += rs[t + st]; rq[t] += rq[t + st]; }
        __syncthreads();
    }
    float mu  = rs[0] * (1.0f / Dc);
    float var = rq[0] * (1.0f / Dc) - mu * mu;
    float inv = rsqrtf(var + 1e-5f);
    _Float16* dst = xn + (size_t)row * Dc;
    for (int i = t; i < Dc; i += 256)
        dst[i] = (_Float16)((xr[i] - mu) * inv * gamma[i] + beta[i]);
}

// ---- f32 -> f16 converter --------------------------------------------------
__global__ void cvt_f16_kernel(const float* __restrict__ src,
                               _Float16* __restrict__ dst, int n) {
    for (int i = blockIdx.x * blockDim.x + threadIdx.x; i < n;
         i += gridDim.x * blockDim.x)
        dst[i] = (_Float16)src[i];
}

// ---- WMMA GEMM: C[M,N] = A[M,K] * B[K,N] -----------------------------------
// Tile staging goes through global_load_async_to_lds_b128 (no VGPR bounce).
// MODE 0: out-proj epilogue -> f32 out + bias
// MODE 1: qkv epilogue -> scatter f16 into q/k/v [B,H,S,Dh], q scaled 1/8
template <int MODE>
__global__ void gemm_wmma_kernel(const _Float16* __restrict__ A,
                                 const _Float16* __restrict__ Bm,
                                 int M, int N, int K,
                                 float* __restrict__ outF,
                                 const float* __restrict__ bias,
                                 _Float16* __restrict__ qh,
                                 _Float16* __restrict__ kh,
                                 _Float16* __restrict__ vh) {
    __shared__ __align__(16) _Float16 As[128][32];
    __shared__ __align__(16) _Float16 Bs[32][128];

    int blockN = blockIdx.x * 128;
    int blockM = blockIdx.y * 128;
    int t    = threadIdx.x;
    int w    = t >> 5;
    int lane = t & 31;
    int hi   = (lane >> 4) & 1;
    int l15  = lane & 15;

    v8f acc[8] = {};

    for (int kt = 0; kt < K; kt += 32) {
        __syncthreads();
        // Stage A (128x32) and B (32x128) tiles; 512 async 16B copies each.
        #pragma unroll
        for (int c = t; c < 512; c += 256) {
            int ra = c >> 2, ca = (c & 3) * 8;
            async_cp16(&As[ra][ca],
                       &A[(size_t)(blockM + ra) * K + kt + ca]);
            int rb = c >> 4, cb = (c & 15) * 8;
            async_cp16(&Bs[rb][cb],
                       &Bm[(size_t)(kt + rb) * N + blockN + cb]);
        }
        if (kt + 32 < K) {  // hint next tiles (global_prefetch_b8)
            __builtin_prefetch(&A[(size_t)(blockM + (t >> 1)) * K + kt + 32]);
            __builtin_prefetch(&Bm[(size_t)(kt + 32 + (t & 31)) * N + blockN]);
        }
        wait_async0();
        __syncthreads();

        v16h a = load_a(&As[w * 16][0], 32);
        #pragma unroll
        for (int nt = 0; nt < 8; ++nt) {
            v16h b = load_b(&Bs[0][nt * 16], 128);
            acc[nt] = wmma_f16(a, b, acc[nt]);
        }
    }

    // Epilogue. C/D layout: vgpr r -> row m = r + 8*hi, lane&15 -> col.
    #pragma unroll
    for (int nt = 0; nt < 8; ++nt) {
        #pragma unroll
        for (int r = 0; r < 8; ++r) {
            int gm = blockM + w * 16 + r + 8 * hi;
            int gn = blockN + nt * 16 + l15;
            float v = acc[nt][r];
            if (MODE == 0) {
                outF[(size_t)gm * N + gn] = v + bias[gn];
            } else {
                int part = gn >> 10;       // 0:q 1:k 2:v
                int nin  = gn & 1023;
                int hh   = nin >> 6;
                int dd   = nin & 63;
                int bb   = gm >> 11;       // row / S
                int ss   = gm & 2047;
                _Float16* dst = (part == 0) ? qh : (part == 1) ? kh : vh;
                float scale = (part == 0) ? 0.125f : 1.0f;  // Dh^-0.5
                dst[(((size_t)(bb * Hc + hh)) * Sc + ss) * Dh + dd] =
                    (_Float16)(v * scale);
            }
        }
    }
}

// ---- Flash attention with block-causal mask --------------------------------
// grid: (S/128, H, B); 256 threads = 8 waves; wave owns 16 query rows.
// 128-row query tiles are aligned inside 256-row windows, so the mask is
// exactly a key-loop bound: keys 0 .. (win+1)*256-1 are fully visible.
// V tiles are DMA'd by the Tensor Data Mover (D# built per ISA 08 §8).
__global__ void attn_wmma_kernel(const _Float16* __restrict__ qh,
                                 const _Float16* __restrict__ kh,
                                 const _Float16* __restrict__ vh,
                                 _Float16* __restrict__ oh) {
    __shared__ __align__(16) _Float16 Kt[64][32];     // K^T tile: [d][j]
    __shared__ __align__(16) _Float16 Vt[32][64];     // V tile:   [j][d]
    __shared__ __align__(16) _Float16 Pb[8][16][32];  // per-wave P bounce

    int qt = blockIdx.x, h = blockIdx.y, b = blockIdx.z;
    int t    = threadIdx.x;
    int w    = t >> 5;
    int lane = t & 31;
    int hi   = lane >> 4;
    int l15  = lane & 15;

    size_t headoff = ((size_t)(b * Hc + h)) * Sc * Dh;
    int qbase = qt * 128 + w * 16;

    // Q fragments straight from global (rows contiguous, stride Dh).
    v16h qa0 = load_a(qh + headoff + (size_t)qbase * Dh, Dh);
    v16h qa1 = load_a(qh + headoff + (size_t)qbase * Dh + 32, Dh);

    v8f o[4] = {};
    float mrun[8], lrun[8];
    #pragma unroll
    for (int r = 0; r < 8; ++r) { mrun[r] = -1e30f; lrun[r] = 0.f; }

    int nkb = ((qt >> 1) + 1) * 8;  // 32-key blocks visible to this window
    for (int kb = 0; kb < nkb; ++kb) {
        int k0 = kb * 32;
        __syncthreads();

        // --- V tile via Tensor Data Mover (wave 0 issues, TENSORcnt waits).
        if (w == 0) {
            unsigned long long ga =
                (unsigned long long)(uintptr_t)(vh + headoff + (size_t)k0 * Dh);
            G0u g0;
            g0.u[0] = 1u;                                 // count=1, no gather
            g0.u[1] = (unsigned)(uintptr_t)(&Vt[0][0]);   // lds_addr
            g0.u[2] = (unsigned)ga;                       // global_addr[31:0]
            g0.u[3] = (unsigned)((ga >> 32) & 0x01FFFFFFu) | (2u << 30); // +type
            G1u g1;
            g1.u[0] = 0x00010000u;       // wg_mask=0, data_size=2B
            g1.u[1] = (64u << 16);       // tensor_dim0 = 64 (Dh)
            g1.u[2] = (2048u << 16);     // tensor_dim1 = S
            g1.u[3] = (64u << 16);       // tile_dim0 = 64
            g1.u[4] = 32u;               // tile_dim1 = 32, tile_dim2 = 0
            g1.u[5] = 64u;               // tensor_dim0_stride = 64
            g1.u[6] = 0u;
            g1.u[7] = 0u;
            asm volatile("tensor_load_to_lds %0, %1"
                         :: "s"(g0.v), "s"(g1.v) : "memory");
            __builtin_amdgcn_s_wait_tensorcnt(0);
        }
        // --- K^T tile: manual transposed scatter (TDM cannot transpose).
        {
            int j  = t >> 3;
            int d0 = (t & 7) * 8;
            h8 kvv = *(const h8*)&kh[headoff + (size_t)(k0 + j) * Dh + d0];
            #pragma unroll
            for (int e = 0; e < 8; ++e) Kt[d0 + e][j] = kvv[e];
        }
        __syncthreads();

        // S = Q * K^T  (two 16-col tiles, contraction Dh=64 in two steps)
        v8f s0 = {}, s1 = {};
        s0 = wmma_f16(qa0, load_b(&Kt[0][0],  32), s0);
        s0 = wmma_f16(qa1, load_b(&Kt[32][0], 32), s0);
        s1 = wmma_f16(qa0, load_b(&Kt[0][16], 32), s1);
        s1 = wmma_f16(qa1, load_b(&Kt[32][16], 32), s1);

        // Online softmax. Row m lives across 16 lanes of a group; xor masks
        // 1/2/4/8 reduce within the group (wave32).
        #pragma unroll
        for (int r = 0; r < 8; ++r) {
            float mx = fmaxf(s0[r], s1[r]);
            mx = fmaxf(mx, __shfl_xor(mx, 1, 32));
            mx = fmaxf(mx, __shfl_xor(mx, 2, 32));
            mx = fmaxf(mx, __shfl_xor(mx, 4, 32));
            mx = fmaxf(mx, __shfl_xor(mx, 8, 32));
            float mnew = fmaxf(mrun[r], mx);
            float corr = __expf(mrun[r] - mnew);
            float p0 = __expf(s0[r] - mnew);
            float p1 = __expf(s1[r] - mnew);
            float rsum = p0 + p1;
            rsum += __shfl_xor(rsum, 1, 32);
            rsum += __shfl_xor(rsum, 2, 32);
            rsum += __shfl_xor(rsum, 4, 32);
            rsum += __shfl_xor(rsum, 8, 32);
            lrun[r] = lrun[r] * corr + rsum;
            mrun[r] = mnew;
            #pragma unroll
            for (int nt = 0; nt < 4; ++nt) o[nt][r] *= corr;
            // C-layout -> A-layout bounce through wave-private LDS strip.
            Pb[w][r + 8 * hi][l15]      = (_Float16)p0;
            Pb[w][r + 8 * hi][16 + l15] = (_Float16)p1;
        }
        __asm__ volatile("" ::: "memory");  // keep ds stores before ds loads

        v16h pa = load_a(&Pb[w][0][0], 32);
        #pragma unroll
        for (int nt = 0; nt < 4; ++nt)
            o[nt] = wmma_f16(pa, load_b(&Vt[0][nt * 16], Dh), o[nt]);
    }

    // Normalize and store O as [B*S, H*Dh] f16 for the out-projection GEMM.
    #pragma unroll
    for (int nt = 0; nt < 4; ++nt) {
        #pragma unroll
        for (int r = 0; r < 8; ++r) {
            int qr  = qbase + r + 8 * hi;
            int col = h * Dh + nt * 16 + l15;
            oh[((size_t)(b * Sc + qr)) * INNER + col] =
                (_Float16)(o[nt][r] / lrun[r]);
        }
    }
}

// ---------------------------------------------------------------------------
extern "C" void kernel_launch(void* const* d_in, const int* in_sizes, int n_in,
                              void* d_out, int out_size, void* d_ws, size_t ws_size,
                              hipStream_t stream) {
    const float* x     = (const float*)d_in[0];
    const float* gamma = (const float*)d_in[1];
    const float* beta  = (const float*)d_in[2];
    const float* Wqkv  = (const float*)d_in[3];
    const float* Wout  = (const float*)d_in[4];
    const float* bout  = (const float*)d_in[5];
    float* out = (float*)d_out;

    _Float16* p = (_Float16*)d_ws;
    _Float16* xnH   = p; p += (size_t)Mrows * Dc;        // 8M
    _Float16* WqkvH = p; p += (size_t)Dc * 3 * INNER;    // 3M
    _Float16* WoutH = p; p += (size_t)INNER * Dc;        // 1M
    _Float16* qH    = p; p += (size_t)Mrows * INNER;     // 8M
    _Float16* kH    = p; p += (size_t)Mrows * INNER;     // 8M
    _Float16* vH    = p; p += (size_t)Mrows * INNER;     // 8M
    _Float16* aH    = p; p += (size_t)Mrows * INNER;     // 8M

    // 1) LayerNorm -> f16
    ln_f16_kernel<<<Mrows, 256, 0, stream>>>(x, gamma, beta, xnH);
    // 2) Weight downcasts
    cvt_f16_kernel<<<2048, 256, 0, stream>>>(Wqkv, WqkvH, Dc * 3 * INNER);
    cvt_f16_kernel<<<1024, 256, 0, stream>>>(Wout, WoutH, INNER * Dc);
    // 3) QKV projection (scatter epilogue, q pre-scaled)
    gemm_wmma_kernel<1><<<dim3(3 * INNER / 128, Mrows / 128), 256, 0, stream>>>(
        xnH, WqkvH, Mrows, 3 * INNER, Dc, nullptr, nullptr, qH, kH, vH);
    // 4) Flash attention, block-causal via loop bound
    attn_wmma_kernel<<<dim3(Sc / 128, Hc, Bc), 256, 0, stream>>>(qH, kH, vH, aH);
    // 5) Output projection + bias -> f32
    gemm_wmma_kernel<0><<<dim3(Dc / 128, Mrows / 128), 256, 0, stream>>>(
        aH, WoutH, Mrows, Dc, INNER, out, bout, nullptr, nullptr, nullptr);
}
